// GMSA_22771916603920
// MI455X (gfx1250) — matrix operand order repaired and checked by reading, not compiled, not run on hardware
//
#include <hip/hip_runtime.h>

// ---------------------------------------------------------------------------
// GMSA for MI455X (gfx1250): fused in-proj(+BN) -> windowed QQ^T softmax V via
// v_wmma_f32_16x16x32_f16 -> out-proj. Attention maps streamed fp32 coalesced.
// V is staged TRANSPOSED in LDS so all WMMA fragment loads are ds_load_b128.
// ---------------------------------------------------------------------------

typedef _Float16 v16h __attribute__((ext_vector_type(16)));
typedef _Float16 v8h  __attribute__((ext_vector_type(8)));
typedef float    v8f  __attribute__((ext_vector_type(8)));

// Two aligned 16-byte LDS loads -> one 16-half fragment register group.
__device__ inline v16h ld2(const _Float16* p0, const _Float16* p1) {
    v8h lo = *(const v8h*)p0;
    v8h hi = *(const v8h*)p1;
    return __builtin_shufflevector(lo, hi, 0, 1, 2, 3, 4, 5, 6, 7,
                                   8, 9, 10, 11, 12, 13, 14, 15);
}

// A-matrix fragment (16x32 f16, M x K), row-major source S[row][k].
// Per ISA: lane L: M=L&15, h=L>>4; e<8 -> K=8h+e ; e>=8 -> K=16+8h+(e-8).
__device__ inline v16h load_A(const _Float16* base, int rowStride, int m0, int k0, int lane) {
    const int M = lane & 15, h = lane >> 4;
    const _Float16* r = base + (size_t)(m0 + M) * rowStride + k0;
    return ld2(r + 8 * h, r + 16 + 8 * h);
}

// B-matrix fragment (32x16 f16, K x N) where B[k][n] = S[n][k] (transposed src).
// Per ISA: lane L: N=L&15, h=L>>4; e<8 -> K=8h+e ; e>=8 -> K=16+8h+(e-8).
__device__ inline v16h load_BT(const _Float16* base, int rowStride, int n0, int k0, int lane) {
    const int N = lane & 15, h = lane >> 4;
    const _Float16* r = base + (size_t)(n0 + N) * rowStride + k0;
    return ld2(r + 8 * h, r + 16 + 8 * h);
}

// ---------------------------------------------------------------------------
// Kernel 0: fold BN into in-projection weights.
// ---------------------------------------------------------------------------
__global__ void fold_weights(const float* __restrict__ W_in, const float* __restrict__ b_in,
                             const float* __restrict__ gamma, const float* __restrict__ beta,
                             const float* __restrict__ mean, const float* __restrict__ var,
                             float* __restrict__ Wf, float* __restrict__ bf) {
    int o = blockIdx.x * blockDim.x + threadIdx.x;
    if (o >= 120) return;
    float inv = gamma[o] * rsqrtf(var[o] + 1e-5f);
    for (int c = 0; c < 60; ++c) Wf[o * 60 + c] = inv * W_in[o * 60 + c];
    bf[o] = b_in[o] * inv + beta[o] - mean[o] * inv;
}

// ---------------------------------------------------------------------------
// Main attention kernel (one instantiation per branch).
//  BR: branch (channels BR*40.. in projected space, BR*20.. in y)
//  WS: window size (4/8/12);  WPB: windows per 256-thread block
// ---------------------------------------------------------------------------
template <int BR, int WS, int WPB>
__launch_bounds__(256)
__global__ void gmsa_attn(const float* __restrict__ x, const float* __restrict__ Wf,
                          const float* __restrict__ bf, float* __restrict__ y_mid,
                          float* __restrict__ atn_out) {
    constexpr int H = 288, Wd = 288, HW = H * Wd, Cin = 60;
    constexpr int P  = WS * WS;              // tokens per window
    constexpr int NH = H / WS, NW = Wd / WS; // windows per image dim
    constexpr int SH = WS / 2;               // roll shift
    constexpr int CP = 32;                   // padded head dim (c=20 -> 32)
    constexpr int KP = ((P + 31) / 32) * 32; // padded token-K for A.V
    constexpr int NT = P / 16;               // 16x16 tiles per dim
    constexpr int OFF = BR * 40;             // projected-channel offset
    constexpr int NWAVES = 8;

    __shared__ __attribute__((aligned(16))) _Float16 qs[WPB][P][CP];   // Q[tok][ch]
    __shared__ __attribute__((aligned(16))) _Float16 vsT[WPB][CP][KP]; // V^T[ch][tok]
    __shared__ __attribute__((aligned(16))) _Float16 lg[WPB][P][KP];   // logits/probs

    const int tid = threadIdx.x;
    const int lane = tid & 31;
    const int wave = tid >> 5;
    const int winBase = blockIdx.x * WPB;

    // ---- Phase 1: fused 1x1 conv + BN  ->  q, v^T in LDS (f16) -------------
    for (int t = tid; t < WPB * P; t += 256) {
        int widx = t / P, tok = t % P;
        int win = winBase + widx;
        int b = win / (NH * NW), rem = win % (NH * NW);
        int wy = rem / NW, wx = rem % NW;
        int row = tok / WS, col = tok % WS;
        int gy = wy * WS + row + SH; if (gy >= H) gy -= H;   // roll by -SH
        int gx = wx * WS + col + SH; if (gx >= Wd) gx -= Wd;
        const float* xp = x + (size_t)b * Cin * HW + (size_t)gy * Wd + gx;
        float acc[40];
#pragma unroll
        for (int o = 0; o < 40; ++o) acc[o] = bf[OFF + o];
        for (int c = 0; c < Cin; ++c) {
            float xv = xp[(size_t)c * HW];
#pragma unroll
            for (int o = 0; o < 40; ++o) acc[o] += Wf[(OFF + o) * Cin + c] * xv;
        }
#pragma unroll
        for (int k = 0; k < CP; ++k) {
            qs[widx][tok][k]  = (k < 20) ? (_Float16)acc[k]      : (_Float16)0.f;
            vsT[widx][k][tok] = (k < 20) ? (_Float16)acc[20 + k] : (_Float16)0.f;
        }
    }
    if constexpr (KP > P) {  // zero-pad V^T token columns so padded-K reads are 0
        for (int i = tid; i < WPB * CP * (KP - P); i += 256) {
            int widx = i / (CP * (KP - P));
            int rem  = i % (CP * (KP - P));
            vsT[widx][rem / (KP - P)][P + rem % (KP - P)] = (_Float16)0.f;
        }
    }
    __syncthreads();

    // ---- Phase 2: logits = Q.Q^T  (one WMMA per 16x16 tile, K=32 padded) ----
    for (int job = wave; job < WPB * NT * NT; job += NWAVES) {
        int widx = job / (NT * NT), t = job % (NT * NT);
        int tm = t / NT, tn = t % NT;
        v16h a  = load_A (&qs[widx][0][0], CP, tm * 16, 0, lane);
        v16h bq = load_BT(&qs[widx][0][0], CP, tn * 16, 0, lane);  // B = Q^T
        v8f acc = {};
        acc = __builtin_amdgcn_wmma_f32_16x16x32_f16(false, a, false, bq,
                                                     (short)0, acc, false, false);
        int M0 = tm * 16 + 8 * (lane >> 4);
        int N_ = tn * 16 + (lane & 15);
#pragma unroll
        for (int j = 0; j < 8; ++j) lg[widx][M0 + j][N_] = (_Float16)acc[j];
    }
    __syncthreads();

    // ---- Phase 3: row softmax (f32 math), overwrite lg with f16 probs ------
    for (int r = tid; r < WPB * P; r += 256) {
        int widx = r / P, rr = r % P;
        _Float16* rowp = &lg[widx][rr][0];
        float mx = -1e30f;
        for (int c = 0; c < P; ++c) mx = fmaxf(mx, (float)rowp[c]);
        float s = 0.f;
        for (int c = 0; c < P; ++c) s += __expf((float)rowp[c] - mx);
        float inv = 1.f / s;
        for (int c = 0; c < P; ++c)
            rowp[c] = (_Float16)(__expf((float)rowp[c] - mx) * inv);
        for (int c = P; c < KP; ++c) rowp[c] = (_Float16)0.f;  // zero K-pad
    }
    __syncthreads();

    // ---- Phase 4: stream attention maps to global (fp32, coalesced) --------
    for (int i = tid; i < WPB * P * P; i += 256) {
        int widx = i / (P * P), o = i % (P * P);
        atn_out[(size_t)winBase * P * P + i] = (float)lg[widx][o / P][o % P];
    }

    // ---- Phase 5: y = atn . V  (WMMA chain over padded token-K) ------------
    for (int job = wave; job < WPB * NT * 2; job += NWAVES) {
        int widx = job / (NT * 2), t = job % (NT * 2);
        int tm = t >> 1, n0 = (t & 1) * 16;
        v8f acc = {};
#pragma unroll
        for (int k0 = 0; k0 < KP; k0 += 32) {
            v16h a  = load_A (&lg[widx][0][0], KP, tm * 16, k0, lane);
            v16h bv = load_BT(&vsT[widx][0][0], KP, n0, k0, lane);  // B[k][n]=vsT[n][k]
            acc = __builtin_amdgcn_wmma_f32_16x16x32_f16(false, a, false, bv,
                                                         (short)0, acc, false, false);
        }
        int win = winBase + widx;
        int b = win / (NH * NW), rem = win % (NH * NW);
        int wy = rem / NW, wx = rem % NW;
        int ch = n0 + (lane & 15);
        if (ch < 20) {
#pragma unroll
            for (int j = 0; j < 8; ++j) {
                int tok = tm * 16 + j + 8 * (lane >> 4);
                int row = tok / WS, col = tok % WS;
                int gy = wy * WS + row + SH; if (gy >= H) gy -= H;  // roll back +SH
                int gx = wx * WS + col + SH; if (gx >= Wd) gx -= Wd;
                y_mid[((size_t)b * 60 + BR * 20 + ch) * HW + (size_t)gy * Wd + gx] = acc[j];
            }
        }
    }
}

// ---------------------------------------------------------------------------
// Kernel: out-projection, 60x60 per pixel, weights scalar-loaded (uniform).
// ---------------------------------------------------------------------------
__launch_bounds__(256)
__global__ void project_out(const float* __restrict__ y_mid, const float* __restrict__ Wout,
                            const float* __restrict__ bout, float* __restrict__ out) {
    constexpr size_t HW = 288 * 288;
    size_t pix = (size_t)blockIdx.x * blockDim.x + threadIdx.x;
    if (pix >= (size_t)8 * HW) return;
    size_t b = pix / HW, hw = pix % HW;
    float acc[60];
#pragma unroll
    for (int o = 0; o < 60; ++o) acc[o] = bout[o];
    for (int c = 0; c < 60; ++c) {
        float xv = y_mid[(b * 60 + c) * HW + hw];
#pragma unroll
        for (int o = 0; o < 60; ++o) acc[o] += Wout[o * 60 + c] * xv;
    }
    for (int o = 0; o < 60; ++o) out[(b * 60 + o) * HW + hw] = acc[o];
}

// ---------------------------------------------------------------------------
extern "C" void kernel_launch(void* const* d_in, const int* in_sizes, int n_in,
                              void* d_out, int out_size, void* d_ws, size_t ws_size,
                              hipStream_t stream) {
    (void)in_sizes; (void)n_in; (void)out_size; (void)ws_size;
    const float* x     = (const float*)d_in[0];
    const float* W_in  = (const float*)d_in[1];
    const float* b_in  = (const float*)d_in[2];
    const float* gamma = (const float*)d_in[3];
    const float* beta  = (const float*)d_in[4];
    const float* mean  = (const float*)d_in[5];
    const float* var   = (const float*)d_in[6];
    const float* W_out = (const float*)d_in[7];
    const float* b_out = (const float*)d_in[8];
    float* out = (float*)d_out;

    // workspace: folded weights/bias + y_mid [8,60,288,288]
    float* Wf    = (float*)d_ws;         // 120*60
    float* bf    = Wf + 120 * 60;        // 128 (padded)
    float* y_mid = bf + 128;             // 8*60*288*288

    constexpr size_t HW = 288 * 288;
    constexpr size_t YE = (size_t)8 * 60 * HW;
    float* atn0 = out + YE;                              // [41472,16,16]
    float* atn1 = atn0 + (size_t)41472 * 16 * 16;        // [10368,64,64]
    float* atn2 = atn1 + (size_t)10368 * 64 * 64;        // [4608,144,144]

    fold_weights<<<1, 128, 0, stream>>>(W_in, b_in, gamma, beta, mean, var, Wf, bf);

    gmsa_attn<0, 4, 8><<<41472 / 8, 256, 0, stream>>>(x, Wf, bf, y_mid, atn0);
    gmsa_attn<1, 8, 1><<<10368,     256, 0, stream>>>(x, Wf, bf, y_mid, atn1);
    gmsa_attn<2, 12, 1><<<4608,     256, 0, stream>>>(x, Wf, bf, y_mid, atn2);

    project_out<<<(int)((8 * HW + 255) / 256), 256, 0, stream>>>(y_mid, W_out, b_out, out);
}
